// MaskedMultiHeadAttention_27367531610189
// MI455X (gfx1250) — matrix operand order repaired and checked by reading, compiled, not run on hardware
//
// MI455X (gfx1250, wave32) masked multi-head attention forward.
// bf16 WMMA (v_wmma_f32_16x16x32_bf16), f32 accumulate everywhere.
// attn [2,16,2048,2048] (536MB > 192MB L2) streamed via b128 NT stores.
// 3-pass softmax (max/sum/write); masking done by additive -1e38 bias
// (no per-element selects, exp underflow gives exact +0 like the reference);
// 1/sqrt(d_k) pre-folded into Q (exact, pow2).  Workspace: 48 MiB of d_ws.
#include <hip/hip_runtime.h>
#include <hip/hip_bf16.h>
#include <float.h>

typedef __bf16 bf16_t;
typedef __attribute__((ext_vector_type(4)))  float  v4f;
typedef __attribute__((ext_vector_type(8)))  float  v8f;
typedef __attribute__((ext_vector_type(8)))  __bf16 v8bf;
typedef __attribute__((ext_vector_type(16))) __bf16 v16bf;

#define D_MODEL   1024
#define NUM_HEADS 16
#define D_K       64
#define BATCH     2
#define SEQ       2048
#define M_TOT     (BATCH*SEQ)   // 4096 rows
#define NEG_BIG   (-1.0e38f)

__device__ __forceinline__ v16bf cat16(v8bf lo, v8bf hi) {
  v16bf r;
#pragma unroll
  for (int i = 0; i < 8; ++i) { r[i] = lo[i]; r[i + 8] = hi[i]; }
  return r;
}

__device__ __forceinline__ v8f v8f_zero() {
  v8f z;
#pragma unroll
  for (int i = 0; i < 8; ++i) z[i] = 0.f;
  return z;
}

__device__ __forceinline__ v8f wmma_bf16(v16bf a, v16bf b, v8f c) {
  // (neg_a, A, neg_b, B, c_mod, C, reuse_a, reuse_b)
  return __builtin_amdgcn_wmma_f32_16x16x32_bf16(false, a, false, b, (short)0, c,
                                                 false, false);
}

// 16-lane half-wave butterflies (xor masks 1,2,4,8 stay within each half).
__device__ __forceinline__ float redmax16(float x) {
  x = fmaxf(x, __shfl_xor(x, 1));
  x = fmaxf(x, __shfl_xor(x, 2));
  x = fmaxf(x, __shfl_xor(x, 4));
  x = fmaxf(x, __shfl_xor(x, 8));
  return x;
}
__device__ __forceinline__ float redsum16(float x) {
  x += __shfl_xor(x, 1);
  x += __shfl_xor(x, 2);
  x += __shfl_xor(x, 4);
  x += __shfl_xor(x, 8);
  return x;
}

// ---------------------------------------------------------------- conversion
__global__ void cvt_f32_to_bf16(const float* __restrict__ src,
                                bf16_t* __restrict__ dst, int n) {
  for (int i = blockIdx.x * blockDim.x + threadIdx.x; i < n;
       i += gridDim.x * blockDim.x)
    dst[i] = (bf16_t)src[i];
}

// ------------------------------------------------------------------- GEMM
// Y = A[M,1024] @ W[N=1024,K=1024]^T  (both bf16, row-major, f32 accumulate)
// MODE 0: store bf16*prescale to [B,H,L,64]   (Q with 0.125, K with 1.0)
// MODE 1: store bf16 to [B,H,64,L]            (V transposed for PV WMMA)
// MODE 2: store f32 + bias to [M,1024]        (final FC -> d_out)
template <int MODE>
__global__ void __launch_bounds__(256)
gemm16(const bf16_t* __restrict__ A, const bf16_t* __restrict__ W,
       void* __restrict__ out, const float* __restrict__ bias, float prescale) {
  const int lane  = threadIdx.x & 31;
  const int wid   = threadIdx.x >> 5;
  const int mtile = blockIdx.x;                 // 0..M/16-1
  const int ntile = blockIdx.y * 8 + wid;       // 0..63
  const int lhalf = lane >> 4;                  // 0|1
  const int lmod  = lane & 15;

  // A fragment: lane half selects K {0..7,16..23} vs {8..15,24..31}
  const bf16_t* aRow = A + (size_t)(mtile * 16 + lmod) * D_MODEL + 8 * lhalf;
  // B fragment: lane half selects K {0..7,16..23} vs {8..15,24..31} of column n (= row n of W)
  const bf16_t* bRow = W + (size_t)(ntile * 16 + lmod) * D_MODEL + 8 * lhalf;

  v8f acc = v8f_zero();
#pragma unroll 4
  for (int k = 0; k < D_MODEL; k += 32) {
    v8bf a0 = *(const v8bf*)(aRow + k);
    v8bf a1 = *(const v8bf*)(aRow + k + 16);
    v8bf b0 = *(const v8bf*)(bRow + k);
    v8bf b1 = *(const v8bf*)(bRow + k + 16);
    acc = wmma_bf16(cat16(a0, a1), cat16(b0, b1), acc);
  }

#pragma unroll
  for (int j = 0; j < 8; ++j) {
    const int m = mtile * 16 + j + 8 * lhalf;   // global row
    const int n = ntile * 16 + lmod;            // global col
    const float v = acc[j];
    if (MODE == 2) {
      ((float*)out)[(size_t)m * D_MODEL + n] = v + bias[n];
    } else {
      const int b = m >> 11, l = m & 2047;      // SEQ = 2048
      const int h = n >> 6,  d = n & 63;        // D_K = 64
      if (MODE == 0)
        ((bf16_t*)out)[((size_t)((b * NUM_HEADS + h) * SEQ) + l) * D_K + d] =
            (bf16_t)(v * prescale);
      else
        ((bf16_t*)out)[((size_t)((b * NUM_HEADS + h) * D_K) + d) * SEQ + l] =
            (bf16_t)v;
    }
  }
}

// --------------------------------------------------------------- attention
// One wave per (b,h,16-query tile).  Q is pre-scaled by 1/sqrt(d_k).
// Pass A: row max   (WMMA + additive mask bias + per-lane max)
// Pass B: row sum   (WMMA + exp; masked columns underflow to exact +0)
// Pass C: exact normalized attn via f32 LDS tile -> b128 NT stores,
//         context = P @ V with WMMA (A-frag rebuilt from the f32 LDS tile).
__global__ void __launch_bounds__(128)
attn_kernel(const bf16_t* __restrict__ Q,   // [B,H,L,64] (prescaled)
            const bf16_t* __restrict__ K,   // [B,H,L,64]
            const bf16_t* __restrict__ Vt,  // [B,H,64,L]
            const int* __restrict__ ids,    // [B,L]
            float* __restrict__ attnOut,    // [B,H,L,L]
            bf16_t* __restrict__ ctx) {     // [B*L, 1024]
  __shared__ __align__(16) float ftile[4][16][32];   // 8 KB: P tile per wave

  const int lane  = threadIdx.x & 31;
  const int w     = threadIdx.x >> 5;
  const int gw    = blockIdx.x * 4 + w;       // 0..4095
  const int qt    = gw & 127;
  const int h     = (gw >> 7) & 15;
  const int b     = gw >> 11;
  const int lhalf = lane >> 4, lmod = lane & 15;
  const int qbase = qt * 16;

  const bf16_t* Qb = Q + ((size_t)((b * NUM_HEADS + h) * SEQ) + qbase + lmod) * D_K;
  const bf16_t* Kb = K + (size_t)((b * NUM_HEADS + h) * SEQ) * D_K;
  const bf16_t* Vb = Vt + (size_t)((b * NUM_HEADS + h) * D_K) * SEQ;
  const int* idrow = ids + b * SEQ;

  // Q fragments: d 0..31 and 32..63
  const v16bf aQ0 = cat16(*(const v8bf*)(Qb + 8 * lhalf),
                          *(const v8bf*)(Qb + 8 * lhalf + 16));
  const v16bf aQ1 = cat16(*(const v8bf*)(Qb + 32 + 8 * lhalf),
                          *(const v8bf*)(Qb + 32 + 8 * lhalf + 16));

  // diagonal-tile causal bias: col>row <=> lmod > j + 8*lhalf  (invariant)
  float dbias[8];
#pragma unroll
  for (int j = 0; j < 8; ++j) dbias[j] = (lmod > (j + 8 * lhalf)) ? NEG_BIG : 0.f;

  // one 16x16 score tile: S = Q_tile @ K_tile^T  (Q prescaled)
  auto qk_tile = [&](int kt) -> v8f {
    const bf16_t* kp = Kb + (size_t)(kt * 16 + lmod) * D_K + 8 * lhalf;
    __builtin_prefetch(kp + 16 * D_K, 0, 3);       // next K tile (WGP scope)
    v16bf bK0 = cat16(*(const v8bf*)(kp), *(const v8bf*)(kp + 16));
    v16bf bK1 = cat16(*(const v8bf*)(kp + 32), *(const v8bf*)(kp + 48));
    return wmma_bf16(aQ1, bK1, wmma_bf16(aQ0, bK0, v8f_zero()));
  };
  // padding bias: one select per tile, applied additively per element
  auto pad_bias = [&](int kt) -> float {
    return (idrow[kt * 16 + lmod] == 0) ? NEG_BIG : 0.f;
  };

  // ---- pass A: per-lane running max (mask-free main loop, diag separate)
  float pmax[8];
#pragma unroll
  for (int j = 0; j < 8; ++j) pmax[j] = NEG_BIG;

  for (int kt = 0; kt < qt; ++kt) {
    v8f c = qk_tile(kt);
    const float pb = pad_bias(kt);
#pragma unroll
    for (int j = 0; j < 8; ++j) pmax[j] = fmaxf(pmax[j], c[j] + pb);
  }
  {  // diagonal tile
    v8f c = qk_tile(qt);
    const float pb = pad_bias(qt);
#pragma unroll
    for (int j = 0; j < 8; ++j) pmax[j] = fmaxf(pmax[j], c[j] + pb + dbias[j]);
  }
  float M[8];
#pragma unroll
  for (int j = 0; j < 8; ++j) M[j] = redmax16(pmax[j]);  // finite: col 0 open

  // ---- pass B: per-lane partial sums of exp(s - M); masked -> exact +0
  float psum[8];
#pragma unroll
  for (int j = 0; j < 8; ++j) psum[j] = 0.f;

  for (int kt = 0; kt < qt; ++kt) {
    v8f c = qk_tile(kt);
    const float pb = pad_bias(kt);
#pragma unroll
    for (int j = 0; j < 8; ++j) psum[j] += __expf(c[j] + pb - M[j]);
  }
  {  // diagonal tile
    v8f c = qk_tile(qt);
    const float pb = pad_bias(qt);
#pragma unroll
    for (int j = 0; j < 8; ++j) psum[j] += __expf(c[j] + pb + dbias[j] - M[j]);
  }
  float inv[8];
#pragma unroll
  for (int j = 0; j < 8; ++j) inv[j] = 1.f / redsum16(psum[j]);

  // ---- pass C: write attn everywhere (zeros past causal), accumulate P@V
  v8f ctxAcc[4];
#pragma unroll
  for (int t = 0; t < 4; ++t) ctxAcc[t] = v8f_zero();

  float* attnBase = attnOut + (size_t)((b * NUM_HEADS + h) * SEQ + qbase) * SEQ;

  for (int pr = 0; pr < 64; ++pr) {             // 32-key pairs cover all 2048
    const int kb = pr * 32;
#pragma unroll
    for (int s2 = 0; s2 < 2; ++s2) {
      const int kt = pr * 2 + s2;
      float pv[8];
      if (kt < qt) {                            // causal interior
        v8f c = qk_tile(kt);
        const float pb = pad_bias(kt);
#pragma unroll
        for (int j = 0; j < 8; ++j)
          pv[j] = __expf(c[j] + pb - M[j]) * inv[j];
      } else if (kt == qt) {                    // diagonal tile
        v8f c = qk_tile(qt);
        const float pb = pad_bias(qt);
#pragma unroll
        for (int j = 0; j < 8; ++j)
          pv[j] = __expf(c[j] + pb + dbias[j] - M[j]) * inv[j];
      } else {                                  // above the diagonal: zeros
#pragma unroll
        for (int j = 0; j < 8; ++j) pv[j] = 0.f;
      }
#pragma unroll
      for (int j = 0; j < 8; ++j)               // D-layout -> f32 LDS tile
        ftile[w][j + 8 * lhalf][s2 * 16 + lmod] = pv[j];
    }

    // coalesced streaming store of the 16x32 tile: 4x b128 NT per lane
    {
      const int r  = lane >> 1;                 // row 0..15
      const int cb = (lane & 1) * 16;           // col half 0|16
      const float* srow = &ftile[w][r][cb];
      float* drow = attnBase + (size_t)r * SEQ + kb + cb;
#pragma unroll
      for (int q4 = 0; q4 < 4; ++q4) {
        v4f t = *(const v4f*)(srow + 4 * q4);
        __builtin_nontemporal_store(t, (v4f*)(drow + 4 * q4));
      }
    }

    if (2 * pr <= qt) {  // pair touches causal region -> contribute to P@V
      // rebuild bf16 A-fragment (16x32 P) from the f32 LDS tile
      const float* prow = &ftile[w][lmod][8 * lhalf];
      v4f f0 = *(const v4f*)(prow);
      v4f f1 = *(const v4f*)(prow + 4);
      v4f f2 = *(const v4f*)(prow + 16);
      v4f f3 = *(const v4f*)(prow + 20);
      v16bf aP;
#pragma unroll
      for (int i = 0; i < 4; ++i) {
        aP[i]      = (bf16_t)f0[i];
        aP[4 + i]  = (bf16_t)f1[i];
        aP[8 + i]  = (bf16_t)f2[i];
        aP[12 + i] = (bf16_t)f3[i];
      }
#pragma unroll
      for (int t = 0; t < 4; ++t) {
        const bf16_t* vp = Vb + (size_t)(t * 16 + lmod) * SEQ + kb + 8 * lhalf;
        __builtin_prefetch(vp + 32, 0, 3);      // next 32-key pair (WGP scope)
        v16bf bV = cat16(*(const v8bf*)(vp), *(const v8bf*)(vp + 16));
        ctxAcc[t] = wmma_bf16(aP, bV, ctxAcc[t]);
      }
    }
  }

  // store context bf16 as [B*L, 1024] rows for the final FC GEMM
#pragma unroll
  for (int t = 0; t < 4; ++t)
#pragma unroll
    for (int j = 0; j < 8; ++j) {
      const int row = qbase + j + 8 * lhalf;
      const int d   = t * 16 + lmod;
      ctx[((size_t)(b * SEQ + row)) * D_MODEL + h * D_K + d] = (bf16_t)ctxAcc[t][j];
    }
}

// ------------------------------------------------------------------ launch
extern "C" void kernel_launch(void* const* d_in, const int* in_sizes, int n_in,
                              void* d_out, int out_size, void* d_ws,
                              size_t ws_size, hipStream_t stream) {
  const float* x    = (const float*)d_in[0];
  const int*   ids  = (const int*)d_in[1];
  const float* wq   = (const float*)d_in[2];
  const float* wk   = (const float*)d_in[3];
  const float* wv   = (const float*)d_in[4];
  const float* wfc  = (const float*)d_in[5];
  const float* bfc  = (const float*)d_in[6];

  float* out  = (float*)d_out;                           // [2,2048,1024]
  float* attn = out + (size_t)M_TOT * D_MODEL;           // [2,16,2048,2048]

  char* ws = (char*)d_ws;                                // 48 MiB used
  bf16_t* xb  = (bf16_t*)(ws);                           //  8 MiB
  bf16_t* wqb = (bf16_t*)(ws + (8u  << 20));             //  2 MiB
  bf16_t* wkb = (bf16_t*)(ws + (10u << 20));
  bf16_t* wvb = (bf16_t*)(ws + (12u << 20));
  bf16_t* wfb = (bf16_t*)(ws + (14u << 20));
  bf16_t* qb  = (bf16_t*)(ws + (16u << 20));             //  8 MiB [B,H,L,64]
  bf16_t* kb  = (bf16_t*)(ws + (24u << 20));             //  8 MiB [B,H,L,64]
  bf16_t* vt  = (bf16_t*)(ws + (32u << 20));             //  8 MiB [B,H,64,L]
  bf16_t* ctx = (bf16_t*)(ws + (40u << 20));             //  8 MiB [B*L,1024]

  cvt_f32_to_bf16<<<4096, 256, 0, stream>>>(x, xb, M_TOT * D_MODEL);
  cvt_f32_to_bf16<<<1024, 256, 0, stream>>>(wq, wqb, D_MODEL * D_MODEL);
  cvt_f32_to_bf16<<<1024, 256, 0, stream>>>(wk, wkb, D_MODEL * D_MODEL);
  cvt_f32_to_bf16<<<1024, 256, 0, stream>>>(wv, wvb, D_MODEL * D_MODEL);
  cvt_f32_to_bf16<<<1024, 256, 0, stream>>>(wfc, wfb, D_MODEL * D_MODEL);

  dim3 g(M_TOT / 16, D_MODEL / 128), blk(256);
  gemm16<0><<<g, blk, 0, stream>>>(xb, wqb, qb, nullptr, 0.125f);  // Q/sqrt(dk)
  gemm16<0><<<g, blk, 0, stream>>>(xb, wkb, kb, nullptr, 1.0f);    // K
  gemm16<1><<<g, blk, 0, stream>>>(xb, wvb, vt, nullptr, 1.0f);    // V (transp)

  attn_kernel<<<1024, 128, 0, stream>>>(qb, kb, vt, ids, attn, ctx);

  gemm16<2><<<g, blk, 0, stream>>>(ctx, wfb, out, bfc, 1.0f);      // FC + bias
}
